// ScaledDotProductAttention_14302241096167
// MI455X (gfx1250) — matrix operand order repaired and checked by reading, not compiled, not run on hardware
//
#include <hip/hip_runtime.h>
#include <hip/hip_bf16.h>

// ---------------------------------------------------------------------------
// MI455X (gfx1250) fused MHA forward.
//   pass 1-3 : Q/K/V projections, 4096x1024x1024 bf16 WMMA GEMM + bias
//              (Q pre-scaled by 1/sqrt(64); V stored transposed [b][h][d][s])
//   pass 4   : flash-attention (online softmax), branchless weight*mask
//              streamed non-temporally (the 670MB roofline term),
//              row-sums computed by an extra WMMA against an all-ones tile
//   pass 5   : output projection, bf16 WMMA GEMM -> fp32 d_out
// ---------------------------------------------------------------------------

typedef __bf16 bf16_t;
typedef __attribute__((ext_vector_type(4)))  float  v4f;
typedef __attribute__((ext_vector_type(8)))  float  v8f;
typedef __attribute__((ext_vector_type(8)))  bf16_t v8bf;
typedef __attribute__((ext_vector_type(16))) bf16_t v16bf;

union V16 { v16bf v; v8bf h[2]; };

#define B_SZ     2
#define H_SZ     16
#define SEQ      2048
#define DMODEL   1024
#define DH       64
#define BIG_NEG  (-3.0e38f)

static __device__ inline v8f wmma_bf16(v16bf a, v16bf b, v8f c) {
  return __builtin_amdgcn_wmma_f32_16x16x32_bf16(false, a, false, b,
                                                 (short)0, c, false, false);
}

// xor-lane swizzle (group-of-32 mode: offset = xor<<10 | and_mask 0x1f)
#define SWZ_XOR(x, m) \
  __int_as_float(__builtin_amdgcn_ds_swizzle(__float_as_int(x), ((m) << 10) | 0x1f))

// ===========================================================================
// GEMM: C[4096,1024] = X[4096,1024] @ W[1024,1024] + bias
//   XBF : 0 = fp32 input (convert to bf16 on tile load), 1 = bf16 input
//   MODE: 0 = store bf16 [b][h][s][64] (with scale)     (Q, K)
//         1 = store bf16 [b][h][64][s] (transposed)     (V)
//         2 = store fp32 row-major -> d_out             (output proj)
// block = 256 threads (8 waves); block tile 128x128; k-step 32
// ===========================================================================
#define LDA 40   // padded LDS row stride (bf16 elems); 80B = 5*16B aligned

template <int XBF, int MODE>
__global__ __launch_bounds__(256)
void gemm_bias_kernel(const void* __restrict__ Xv, const float* __restrict__ W,
                      const float* __restrict__ bias, void* __restrict__ outv,
                      float scale) {
  __shared__ bf16_t As[128 * LDA];   // [m][k]
  __shared__ bf16_t Bs[128 * LDA];   // [n][k]  (W tile transposed)

  const int tid  = threadIdx.x;
  const int lane = tid & 31;
  const int wid  = tid >> 5;
  const int wm   = wid & 3;          // 4 wave-rows  of 32
  const int wn   = wid >> 2;         // 2 wave-cols  of 64
  const int m0   = blockIdx.y * 128;
  const int n0   = blockIdx.x * 128;

  v8f acc[2][4] = {};

  for (int k0 = 0; k0 < DMODEL; k0 += 32) {
    // ---- X tile 128x32 -> As (two b128 LDS stores per thread) -----------
    {
      const int r = tid >> 1;
      const int c = (tid & 1) * 16;
      V16 u;
      if (XBF) {
        const bf16_t* src = (const bf16_t*)Xv + (size_t)(m0 + r) * DMODEL + k0 + c;
        u.h[0] = *(const v8bf*)src;
        u.h[1] = *(const v8bf*)(src + 8);
      } else {
        const float* src = (const float*)Xv + (size_t)(m0 + r) * DMODEL + k0 + c;
#pragma unroll
        for (int j = 0; j < 4; ++j) {
          v4f x = *(const v4f*)(src + j * 4);
#pragma unroll
          for (int e = 0; e < 4; ++e) u.v[j * 4 + e] = (bf16_t)x[e];
        }
      }
      *(v8bf*)&As[r * LDA + c]     = u.h[0];
      *(v8bf*)&As[r * LDA + c + 8] = u.h[1];
    }
    // ---- W tile 32x128 -> Bs (transposed; scalar b16 stores) ------------
    {
      const int kk = tid >> 3;
      const int nn = (tid & 7) * 16;
      const float* src = W + (size_t)(k0 + kk) * DMODEL + n0 + nn;
#pragma unroll
      for (int j = 0; j < 4; ++j) {
        v4f x = *(const v4f*)(src + j * 4);
#pragma unroll
        for (int e = 0; e < 4; ++e) Bs[(nn + j * 4 + e) * LDA + kk] = (bf16_t)x[e];
      }
    }
    __syncthreads();

    // ---- fragments (ISA 16x16x32 bf16 layouts) --------------------------
    v16bf af[2], bfr[4];
    {
      const int r  = lane & 15;
      const int kh = (lane >> 4) * 8;          // A: halves split K 0-7/8-15
#pragma unroll
      for (int i = 0; i < 2; ++i) {
        const bf16_t* p = &As[(wm * 32 + i * 16 + r) * LDA + kh];
        V16 u; u.h[0] = *(const v8bf*)p; u.h[1] = *(const v8bf*)(p + 16);
        af[i] = u.v;
      }
      const int n  = lane & 15;
      const int kb = (lane >> 4) * 8;          // B: halves split K 0-7/8-15
#pragma unroll
      for (int j = 0; j < 4; ++j) {
        const bf16_t* p = &Bs[(wn * 64 + j * 16 + n) * LDA + kb];
        V16 u; u.h[0] = *(const v8bf*)p; u.h[1] = *(const v8bf*)(p + 16);
        bfr[j] = u.v;
      }
    }
#pragma unroll
    for (int i = 0; i < 2; ++i)
#pragma unroll
      for (int j = 0; j < 4; ++j)
        acc[i][j] = wmma_bf16(af[i], bfr[j], acc[i][j]);
    __syncthreads();
  }

  // ---- epilogue: bias + layout-specific store ---------------------------
  const int rHi = (lane >> 4) * 8;   // C layout: lanes 16-31 hold rows r+8
  const int cN  = lane & 15;
#pragma unroll
  for (int i = 0; i < 2; ++i) {
#pragma unroll
    for (int j = 0; j < 4; ++j) {
      const int col = n0 + wn * 64 + j * 16 + cN;
      const float bv = bias[col];
#pragma unroll
      for (int r = 0; r < 8; ++r) {
        const int row = m0 + wm * 32 + i * 16 + r + rHi;
        const float val = acc[i][j][r] + bv;
        if (MODE == 2) {
          ((float*)outv)[(size_t)row * DMODEL + col] = val;
        } else {
          const int b = row >> 11, s = row & (SEQ - 1);
          const int h = col >> 6,  d = col & (DH - 1);
          const bf16_t o = (bf16_t)(val * scale);
          if (MODE == 0)
            ((bf16_t*)outv)[(((size_t)(b * H_SZ + h) * SEQ) + s) * DH + d] = o;
          else   // V transposed: [b][h][d][s]
            ((bf16_t*)outv)[(((size_t)(b * H_SZ + h) * DH) + d) * SEQ + s] = o;
        }
      }
    }
  }
}

// ===========================================================================
// Flash attention. 256 threads = 8 waves; each wave owns one 16-row q-tile
// and streams keys in chunks of 32 (two 16x16 score tiles per chunk).
// __launch_bounds__(256,1): latency here is hidden by loads in flight, not
// wave count -- allow the allocator to keep the whole w/mask stream resident.
// ===========================================================================
__global__ __launch_bounds__(256, 1)
void attention_kernel(const bf16_t* __restrict__ Q, const bf16_t* __restrict__ K,
                      const bf16_t* __restrict__ Vt, const float* __restrict__ attw,
                      const unsigned char* __restrict__ mask,
                      bf16_t* __restrict__ ctx) {
  __shared__ bf16_t p_lds[8][16][32];          // per-wave P staging (C->A relayout)

  const int lane = threadIdx.x & 31;
  const int wid  = threadIdx.x >> 5;
  const int b    = blockIdx.z;
  const int h    = blockIdx.y;
  const int q0   = (blockIdx.x * 8 + wid) * 16;
  const int bh   = b * H_SZ + h;

  // ---- Q fragments (A layout, d split 0-31 / 32-63) ---------------------
  v16bf aq[2];
  {
    const int r = lane & 15, kh = (lane >> 4) * 8;
    const bf16_t* qp = Q + ((size_t)bh * SEQ + q0) * DH + r * DH;
#pragma unroll
    for (int c = 0; c < 2; ++c) {
      const bf16_t* p = qp + c * 32 + kh;
      V16 u; u.h[0] = *(const v8bf*)p; u.h[1] = *(const v8bf*)(p + 16);
      aq[c] = u.v;
    }
  }

  // all-ones B fragment: row sums of P via one extra WMMA
  v16bf ones;
#pragma unroll
  for (int e = 0; e < 16; ++e) ones[e] = (bf16_t)1.0f;

  float mrow[8], lrow[8];
#pragma unroll
  for (int r = 0; r < 8; ++r) { mrow[r] = BIG_NEG; lrow[r] = 0.f; }
  v8f o[4] = {};

  const float*         wbase = attw + (size_t)bh * SEQ * SEQ;
  const unsigned char* mbase = mask + (size_t)bh * SEQ * SEQ;
  const int rHi = (lane >> 4) * 8;
  const int nC  = lane & 15;

  for (int kb = 0; kb < SEQ; kb += 32) {
    // ---- stream weights + mask for this 16x32 tile (branchless) ---------
    float wv[2][8], pen[2][8];
#pragma unroll
    for (int t = 0; t < 2; ++t)
#pragma unroll
      for (int r = 0; r < 8; ++r) {
        const size_t idx = (size_t)(q0 + r + rHi) * SEQ + kb + t * 16 + nC;
        if (t == 0 && r == 0) {                // prefetch next 32-key chunk
          __builtin_prefetch(wbase + idx + 32, 0, 0);
          __builtin_prefetch(mbase + idx + 32, 0, 0);
        }
        wv[t][r]  = __builtin_nontemporal_load(wbase + idx);
        pen[t][r] = (float)__builtin_nontemporal_load(mbase + idx) * BIG_NEG;
      }
    // keep the whole stream batch issued ahead of the math; nothing from the
    // FMA chain may be hoisted above, nothing from the batch may sink below
    __builtin_amdgcn_sched_barrier(0);

    // ---- S = Qs @ K^T ----------------------------------------------------
    v8f s[2];
#pragma unroll
    for (int t = 0; t < 2; ++t) {
      v16bf bk[2];
      {
        const int kh = (lane >> 4) * 8;
        const bf16_t* kp = K + ((size_t)bh * SEQ + kb + t * 16 + nC) * DH;
#pragma unroll
        for (int c = 0; c < 2; ++c) {
          const bf16_t* p = kp + c * 32 + kh;
          V16 u; u.h[0] = *(const v8bf*)p; u.h[1] = *(const v8bf*)(p + 16);
          bk[c] = u.v;
        }
      }
      v8f sa = {};
      sa = wmma_bf16(aq[0], bk[0], sa);
      sa = wmma_bf16(aq[1], bk[1], sa);
      // att = S*w + (mask ? -3e38 : 0)   -- pure arithmetic, no branches
#pragma unroll
      for (int r = 0; r < 8; ++r) sa[r] = __builtin_fmaf(sa[r], wv[t][r], pen[t][r]);
      s[t] = sa;
    }

    // ---- online softmax: row max via ds_swizzle xor-reduce --------------
    float f[8];
#pragma unroll
    for (int r = 0; r < 8; ++r) {
      float tm = fmaxf(s[0][r], s[1][r]);
      tm = fmaxf(tm, SWZ_XOR(tm, 1));
      tm = fmaxf(tm, SWZ_XOR(tm, 2));
      tm = fmaxf(tm, SWZ_XOR(tm, 4));
      tm = fmaxf(tm, SWZ_XOR(tm, 8));
      const float nm = fmaxf(mrow[r], tm);
      f[r] = __expf(mrow[r] - nm);
      s[0][r] = __expf(s[0][r] - nm);
      s[1][r] = __expf(s[1][r] - nm);
      mrow[r] = nm;
    }
#pragma unroll
    for (int j = 0; j < 4; ++j)
#pragma unroll
      for (int r = 0; r < 8; ++r) o[j][r] *= f[r];

    // ---- P: C layout -> LDS -> A fragment (same-wave DS ops in-order) ---
#pragma unroll
    for (int t = 0; t < 2; ++t)
#pragma unroll
      for (int r = 0; r < 8; ++r)
        p_lds[wid][r + rHi][t * 16 + nC] = (bf16_t)s[t][r];

    v16bf pa;
    {
      const int r = lane & 15, kh = (lane >> 4) * 8;
      const bf16_t* p = &p_lds[wid][r][kh];
      V16 u; u.h[0] = *(const v8bf*)p; u.h[1] = *(const v8bf*)(p + 16);
      pa = u.v;
    }

    // ---- row sums of P in one WMMA (replaces 32 bpermutes) --------------
    v8f sums = {};
    sums = wmma_bf16(pa, ones, sums);
#pragma unroll
    for (int r = 0; r < 8; ++r) lrow[r] = __builtin_fmaf(lrow[r], f[r], sums[r]);

    // ---- O += P @ V : V^T rows are contiguous along keys ----------------
    const int khv = (lane >> 4) * 8;
#pragma unroll
    for (int j = 0; j < 4; ++j) {
      const bf16_t* vp = Vt + ((size_t)bh * DH + j * 16 + nC) * SEQ + kb + khv;
      V16 u; u.h[0] = *(const v8bf*)vp; u.h[1] = *(const v8bf*)(vp + 16);
      o[j] = wmma_bf16(pa, u.v, o[j]);
    }
  }

  // ---- finalize: O /= l, store context [b][s][h*64+d] bf16 --------------
#pragma unroll
  for (int j = 0; j < 4; ++j)
#pragma unroll
    for (int r = 0; r < 8; ++r) {
      const float val = o[j][r] / lrow[r];
      const size_t row = (size_t)(b * SEQ + q0 + r + rHi);
      ctx[row * DMODEL + h * DH + j * 16 + nC] = (bf16_t)val;
    }
}

// ===========================================================================
extern "C" void kernel_launch(void* const* d_in, const int* in_sizes, int n_in,
                              void* d_out, int out_size, void* d_ws, size_t ws_size,
                              hipStream_t stream) {
  const float*         queries = (const float*)d_in[0];
  const float*         keys    = (const float*)d_in[1];
  const float*         values  = (const float*)d_in[2];
  const float*         attw    = (const float*)d_in[3];
  const unsigned char* amask   = (const unsigned char*)d_in[4];
  const float* Wq = (const float*)d_in[5];  const float* bq = (const float*)d_in[6];
  const float* Wk = (const float*)d_in[7];  const float* bk = (const float*)d_in[8];
  const float* Wv = (const float*)d_in[9];  const float* bv = (const float*)d_in[10];
  const float* Wo = (const float*)d_in[11]; const float* bo = (const float*)d_in[12];
  float* out = (float*)d_out;

  const size_t ELEMS = (size_t)B_SZ * H_SZ * SEQ * DH;   // 4M bf16 each
  bf16_t* wsQ = (bf16_t*)d_ws;
  bf16_t* wsK = wsQ + ELEMS;
  bf16_t* wsV = wsK + ELEMS;
  bf16_t* wsC = wsV + ELEMS;                              // 32 MB total

  const dim3 gGemm(DMODEL / 128, (B_SZ * SEQ) / 128);     // 8 x 32 blocks
  const dim3 gAttn(SEQ / 128, H_SZ, B_SZ);                // 16 x 16 x 2

  // Q projection: fold 1/sqrt(64) into Q
  gemm_bias_kernel<0, 0><<<gGemm, 256, 0, stream>>>(queries, Wq, bq, wsQ, 0.125f);
  gemm_bias_kernel<0, 0><<<gGemm, 256, 0, stream>>>(keys,    Wk, bk, wsK, 1.0f);
  gemm_bias_kernel<0, 1><<<gGemm, 256, 0, stream>>>(values,  Wv, bv, wsV, 1.0f);

  attention_kernel<<<gAttn, 256, 0, stream>>>(wsQ, wsK, wsV, attw, amask, wsC);

  gemm_bias_kernel<1, 2><<<gGemm, 256, 0, stream>>>(wsC, Wo, bo, out, 1.0f);
}
